// CustomQuantizedLinear_56934086476460
// MI455X (gfx1250) — compile-verified, matrix operand order checked by reading, not run on hardware
//
#include <hip/hip_runtime.h>

// ---------------------------------------------------------------------------
// CustomQuantizedLinear for MI455X (gfx1250, wave32, WMMA)
//   out = dequant( int8(x) @ int8(w)^T ) * (sx*sw) + bias
// GEMM core: v_wmma_i32_16x16x64_iu8, 128x128 block tile, k-step 128
// (2 x 64-K chunks per LDS stage -> 16 WMMA/wave between barriers).
// Tile staging: GLOBAL_LOAD_ASYNC_TO_LDS_B128 (ASYNCcnt) double buffering —
// no VGPR round trip; async engine fills next buffer while WMMAs consume
// the current one.
// ---------------------------------------------------------------------------

typedef __attribute__((ext_vector_type(8))) int v8i;
typedef __attribute__((ext_vector_type(4))) int v4i;

#define MDIM 8192
#define NDIM 4096
#define KDIM 4096

#define BM 128
#define BN 128
#define BK 128
#define LDSP (BK + 16)   // padded LDS row stride (bytes) to spread banks

// ---------------------------------------------------------------- init ws ---
__global__ __launch_bounds__(32)
void init_absmax(unsigned* sbits) {
    if (threadIdx.x < 2) sbits[threadIdx.x] = 0u;
}

// -------------------------------------------------------------- abs-max -----
__global__ __launch_bounds__(256)
void absmax_kernel(const float* __restrict__ in, long n, unsigned* __restrict__ out) {
    __shared__ unsigned red[256];
    float m = 0.0f;
    for (long i = (long)blockIdx.x * blockDim.x + threadIdx.x; i < n;
         i += (long)gridDim.x * blockDim.x) {
        m = fmaxf(m, fabsf(in[i]));
    }
    red[threadIdx.x] = __float_as_uint(m);   // bit pattern of |x| is monotone
    __syncthreads();
    for (int s = 128; s > 0; s >>= 1) {
        if (threadIdx.x < s) {
            unsigned a = red[threadIdx.x], b = red[threadIdx.x + s];
            red[threadIdx.x] = a > b ? a : b;
        }
        __syncthreads();
    }
    if (threadIdx.x == 0) atomicMax(out, red[0]);
}

// ------------------------------------------------------------- quantize -----
// 4 floats -> 1 packed int32 of 4 int8 per thread
__global__ __launch_bounds__(256)
void quantize_kernel(const float* __restrict__ in, long n4,
                     const unsigned* __restrict__ absmax_bits,
                     int* __restrict__ outq) {
    long i = (long)blockIdx.x * blockDim.x + threadIdx.x;
    if (i >= n4) return;
    float amax = __uint_as_float(*absmax_bits);
    float inv  = amax > 0.0f ? (127.0f / amax) : 0.0f;
    float4 v = ((const float4*)in)[i];
    int a = __float2int_rn(v.x * inv);
    int b = __float2int_rn(v.y * inv);
    int c = __float2int_rn(v.z * inv);
    int d = __float2int_rn(v.w * inv);
    a = min(127, max(-127, a));
    b = min(127, max(-127, b));
    c = min(127, max(-127, c));
    d = min(127, max(-127, d));
    outq[i] = (a & 0xff) | ((b & 0xff) << 8) | ((c & 0xff) << 16) | ((d & 0xff) << 24);
}

// ----------------------------------------------------------------- GEMM -----
// Block: 256 threads = 8 waves (wave32). Block tile 128x128, k-step 128.
// Wave grid 2(M) x 4(N): each wave -> 4x2 tiles of 16x16 -> 16 WMMA / stage.
// Per-lane fragment load: lane l holds row (l&15); lanes<16 take K 0..31 of a
// 64-chunk, lanes>=16 take K 32..63 (consistent K-permutation on A and B
// => exact dot product, no shuffling needed).
__global__ __launch_bounds__(256)
void gemm_i8_wmma(const char* __restrict__ xq, const char* __restrict__ wq,
                  const unsigned* __restrict__ sbits,
                  const float* __restrict__ bias, float* __restrict__ out) {
    __shared__ char ldsA[2][BM * LDSP];
    __shared__ char ldsB[2][BN * LDSP];

    const int tid  = threadIdx.x;
    const int lane = tid & 31;
    const int wave = tid >> 5;
    const int wm   = wave >> 2;                 // 0..1 -> 64-row slab
    const int wn   = wave & 3;                  // 0..3 -> 32-col slab
    const int blockN0 = blockIdx.x * BN;
    const int blockM0 = blockIdx.y * BM;

    // cooperative tile fill: each thread moves 64 bytes per tile via
    // 4 x global_load_async_to_lds_b128 (INST_OFFSET advances LDS and global)
    const int ldRow = tid >> 1;                 // 0..127
    const int ldCol = (tid & 1) * 64;           // 0 or 64
    const char* gA = xq + (long)(blockM0 + ldRow) * KDIM + ldCol;
    const char* gB = wq + (long)(blockN0 + ldRow) * KDIM + ldCol;

    // low 32 bits of a flat shared pointer == LDS byte offset
    const unsigned ldsa0 = (unsigned)(uintptr_t)&ldsA[0][ldRow * LDSP + ldCol];
    const unsigned ldsb0 = (unsigned)(uintptr_t)&ldsB[0][ldRow * LDSP + ldCol];
    constexpr unsigned BUFSZ = (unsigned)(BM * LDSP);   // 18432 bytes

    auto issue_async = [&](int buf, int step) {
        const char* ga = gA + (long)step * BK;
        const char* gb = gB + (long)step * BK;
        const unsigned la = ldsa0 + (unsigned)buf * BUFSZ;
        const unsigned lb = ldsb0 + (unsigned)buf * BUFSZ;
        asm volatile(
            "global_load_async_to_lds_b128 %0, %2, off\n\t"
            "global_load_async_to_lds_b128 %0, %2, off offset:16\n\t"
            "global_load_async_to_lds_b128 %0, %2, off offset:32\n\t"
            "global_load_async_to_lds_b128 %0, %2, off offset:48\n\t"
            "global_load_async_to_lds_b128 %1, %3, off\n\t"
            "global_load_async_to_lds_b128 %1, %3, off offset:16\n\t"
            "global_load_async_to_lds_b128 %1, %3, off offset:32\n\t"
            "global_load_async_to_lds_b128 %1, %3, off offset:48"
            :: "v"(la), "v"(lb), "v"(ga), "v"(gb)
            : "memory");
    };

    v8i acc[4][2];
    #pragma unroll
    for (int mi = 0; mi < 4; ++mi)
        #pragma unroll
        for (int ni = 0; ni < 2; ++ni)
            acc[mi][ni] = (v8i){0, 0, 0, 0, 0, 0, 0, 0};

    // fill buffer 0
    issue_async(0, 0);
    asm volatile("s_wait_asynccnt 0x0" ::: "memory");
    __syncthreads();

    const int halfsel = (lane >> 4) * 32;       // K 0..31 vs 32..63 of a chunk
    const int rloc    = lane & 15;

    union Frag { v8i v; v4i h[2]; };

    const int nsteps = KDIM / BK;               // 32
    for (int step = 0; step < nsteps; ++step) {
        const int buf = step & 1;
        // async engine fills the other buffer while we do WMMA on this one;
        // safe: the barrier ending step-1 guaranteed all reads of buf^1 done
        if (step + 1 < nsteps) issue_async(buf ^ 1, step + 1);

        #pragma unroll
        for (int c = 0; c < 2; ++c) {           // two 64-K chunks per stage
            const int koff = c * 64 + halfsel;

            Frag a[4], b[2];
            #pragma unroll
            for (int mi = 0; mi < 4; ++mi) {
                const char* p = &ldsA[buf][(wm * 64 + mi * 16 + rloc) * LDSP + koff];
                a[mi].h[0] = *(const v4i*)(p);
                a[mi].h[1] = *(const v4i*)(p + 16);
            }
            #pragma unroll
            for (int ni = 0; ni < 2; ++ni) {
                const char* p = &ldsB[buf][(wn * 32 + ni * 16 + rloc) * LDSP + koff];
                b[ni].h[0] = *(const v4i*)(p);
                b[ni].h[1] = *(const v4i*)(p + 16);
            }

            #pragma unroll
            for (int mi = 0; mi < 4; ++mi)
                #pragma unroll
                for (int ni = 0; ni < 2; ++ni)
                    acc[mi][ni] = __builtin_amdgcn_wmma_i32_16x16x64_iu8(
                        /*sgn_a=*/true, a[mi].v, /*sgn_b=*/true, b[ni].v,
                        acc[mi][ni], /*reuse_a=*/false, /*reuse_b=*/false);
        }

        if (step + 1 < nsteps)
            asm volatile("s_wait_asynccnt 0x0" ::: "memory");
        __syncthreads();
    }

    // fused dequant + bias epilogue
    const float ax = __uint_as_float(sbits[0]);
    const float aw = __uint_as_float(sbits[1]);
    const float scale = (ax * aw) * (1.0f / (127.0f * 127.0f));

    #pragma unroll
    for (int mi = 0; mi < 4; ++mi) {
        #pragma unroll
        for (int ni = 0; ni < 2; ++ni) {
            const int ncol  = blockN0 + wn * 32 + ni * 16 + (lane & 15);
            const int mbase = blockM0 + wm * 64 + mi * 16 + (lane >> 4) * 8;
            const float bv = bias[ncol];
            #pragma unroll
            for (int r = 0; r < 8; ++r) {
                out[(long)(mbase + r) * NDIM + ncol] =
                    (float)acc[mi][ni][r] * scale + bv;
            }
        }
    }
}

// ---------------------------------------------------------------- launch ----
extern "C" void kernel_launch(void* const* d_in, const int* in_sizes, int n_in,
                              void* d_out, int out_size, void* d_ws, size_t ws_size,
                              hipStream_t stream) {
    const float* x    = (const float*)d_in[0];   // [M, K] fp32
    const float* w    = (const float*)d_in[1];   // [N, K] fp32
    const float* bias = (const float*)d_in[2];   // [N]    fp32
    float* out = (float*)d_out;                  // [M, N] fp32

    unsigned char* ws = (unsigned char*)d_ws;
    unsigned* sbits = (unsigned*)ws;                          // [0]=amax(x), [1]=amax(w)
    char* xq = (char*)(ws + 256);                             // M*K int8
    char* wq = (char*)(ws + 256 + (size_t)MDIM * KDIM);       // N*K int8

    init_absmax<<<1, 32, 0, stream>>>(sbits);

    absmax_kernel<<<1024, 256, 0, stream>>>(x, (long)MDIM * KDIM, &sbits[0]);
    absmax_kernel<<<1024, 256, 0, stream>>>(w, (long)NDIM * KDIM, &sbits[1]);

    const long nx4 = (long)MDIM * KDIM / 4;
    const long nw4 = (long)NDIM * KDIM / 4;
    quantize_kernel<<<(int)(nx4 / 256), 256, 0, stream>>>(x, nx4, &sbits[0], (int*)xq);
    quantize_kernel<<<(int)(nw4 / 256), 256, 0, stream>>>(w, nw4, &sbits[1], (int*)wq);

    dim3 grid(NDIM / BN, MDIM / BM);   // 32 x 64
    gemm_i8_wmma<<<grid, 256, 0, stream>>>(xq, wq, sbits, bias, out);
}
